// multiHeadAttention_16862041604187
// MI455X (gfx1250) — matrix operand order repaired and checked by reading, not compiled, not run on hardware
//
#include <hip/hip_runtime.h>

// ---------------------------------------------------------------------------
// MI455X (gfx1250) multi-head attention forward, bf16 WMMA + flash attention.
//
// Roofline: ~137 GFLOP total vs ~200MB HBM traffic -> strongly compute bound
// -> every matmul goes through v_wmma_f32_16x16x32_bf16 (wave32, 16x16 tiles,
// f32 accumulation). The S x S score matrix is never materialized (online
// softmax in registers/LDS). GEMM tile staging is software-pipelined through
// the gfx1250 async global->LDS DMA path: double-buffered LDS, issue tile i+1
// then s_wait_asynccnt 4 (ASYNCcnt completes in order) so the DMA of the next
// tile overlaps the WMMAs of the current one.
// ---------------------------------------------------------------------------

typedef __attribute__((ext_vector_type(16))) __bf16 v16bf;
typedef __attribute__((ext_vector_type(8)))  float  v8f;

#if __has_builtin(__builtin_amdgcn_global_load_async_to_lds_b128) && \
    __has_builtin(__builtin_amdgcn_s_wait_asynccnt)
#define USE_ASYNC_LDS 1
#else
#define USE_ASYNC_LDS 0
#endif

#if USE_ASYNC_LDS
// Exact parameter types per this toolchain's diagnostic:
//   param0: int __vector_size__(16) AS1*  (global source)
//   param1: (by symmetry)           AS3*  (LDS destination)
typedef int v4i __attribute__((vector_size(16)));
typedef __attribute__((address_space(1))) v4i g_v4i;
typedef __attribute__((address_space(3))) v4i l_v4i;

__device__ __forceinline__ void async_ld16(const unsigned short* g,
                                           unsigned short* l) {
    __builtin_amdgcn_global_load_async_to_lds_b128((g_v4i*)g, (l_v4i*)l, 0, 0);
}
#endif

union Frag {            // 16 bf16 = 8 dwords; filled via two 16B loads
    v16bf v;
    uint4 q[2];
    unsigned short h[16];
};

__device__ __forceinline__ unsigned short f32_to_bf16(float f) {
    unsigned int u = __float_as_uint(f);
    unsigned int r = u + 0x7FFFu + ((u >> 16) & 1u);   // round-to-nearest-even
    return (unsigned short)(r >> 16);
}
__device__ __forceinline__ float bf16_to_f32(unsigned short h) {
    return __uint_as_float(((unsigned int)h) << 16);
}

__device__ __forceinline__ v8f wmma_bf16(const Frag& a, const Frag& b, v8f c) {
    return __builtin_amdgcn_wmma_f32_16x16x32_bf16(
        false, a.v, false, b.v, (short)0, c, false, false);
}

// ds_swizzle_b32 xor-shuffle: group-of-32 mode, offset = xor<<10 | and_mask.
// No index VGPR setup (vs ds_bpermute from __shfl_xor).
template <int XOR>
__device__ __forceinline__ float swz_xor(float x) {
    return __int_as_float(__builtin_amdgcn_ds_swizzle(
        __float_as_int(x), (XOR << 10) | 0x1F));
}
__device__ __forceinline__ float half_reduce_max(float x) {
    x = fmaxf(x, swz_xor<1>(x));
    x = fmaxf(x, swz_xor<2>(x));
    x = fmaxf(x, swz_xor<4>(x));
    x = fmaxf(x, swz_xor<8>(x));
    return x;   // reduced across each 16-lane half (rows stay per-half)
}
__device__ __forceinline__ float half_reduce_sum(float x) {
    x += swz_xor<1>(x);
    x += swz_xor<2>(x);
    x += swz_xor<4>(x);
    x += swz_xor<8>(x);
    return x;
}

// ------------------------- f32 -> bf16 conversion --------------------------
__global__ void cvt_f32_bf16_kernel(const float* __restrict__ src,
                                    unsigned short* __restrict__ dst, int n) {
    int i = blockIdx.x * blockDim.x + threadIdx.x;
    if (i < n) dst[i] = f32_to_bf16(src[i]);
}

// ------------------------------- GEMM --------------------------------------
// C[M=8192][N=1024] = A(bf16, M x 1024 row-major) @ Bw(bf16, 1024 x 1024
// row-major)^T.  Both operands contract over their last dim (NT layout),
// which maps directly onto the WMMA A (MxK) / B (KxN with B[c][n]=W[n][c])
// fragment layouts.
// MODE 0: write bf16, scattered into (B=4, H=16, S=2048, D=64) layout.
// MODE 1: write f32 row-major + bias (final projection -> d_out).
template <int MODE>
__global__ __launch_bounds__(256) void gemm_bf16_kernel(
    const unsigned short* __restrict__ A,
    const unsigned short* __restrict__ Bw,
    unsigned short* __restrict__ dstbf,
    float* __restrict__ dstf,
    const float* __restrict__ bias) {
    __shared__ unsigned short As[2][128 * 32];
    __shared__ unsigned short Bs[2][128 * 32];

    const int tid  = threadIdx.x;
    const int wid  = tid >> 5;
    const int lane = tid & 31;
    const int l16  = lane & 15;
    const int hi   = lane >> 4;

    const int m0 = blockIdx.y * 128;
    const int n0 = blockIdx.x * 128;
    const int wm = wid >> 1;   // 0..3  (32 rows each)
    const int wn = wid & 1;    // 0..1  (64 cols each)

    v8f acc[2][4];
#pragma unroll
    for (int i = 0; i < 2; ++i)
#pragma unroll
        for (int j = 0; j < 4; ++j)
#pragma unroll
            for (int e = 0; e < 8; ++e) acc[i][j][e] = 0.0f;

    const int lr = tid >> 1;            // 0..127: tile row this thread loads
    const int lc = (tid & 1) * 16;      // 0/16: 16-half chunk within BK=32
    const unsigned short* gA = A + (size_t)(m0 + lr) * 1024 + lc;
    const unsigned short* gB = Bw + (size_t)(n0 + lr) * 1024 + lc;

    for (int i = 0; i < 32; ++i) {      // 32 k-steps of BK=32
#if USE_ASYNC_LDS
        const int cur = i & 1;
        // Top barrier: nobody still reads the buffer the new DMAs target.
        __syncthreads();
        if (i == 0) {
            async_ld16(gA,     &As[0][lr * 32 + lc]);
            async_ld16(gA + 8, &As[0][lr * 32 + lc + 8]);
            async_ld16(gB,     &Bs[0][lr * 32 + lc]);
            async_ld16(gB + 8, &Bs[0][lr * 32 + lc + 8]);
        }
        if (i + 1 < 32) {
            const int kn = (i + 1) * 32;
            async_ld16(gA + kn,     &As[1 - cur][lr * 32 + lc]);
            async_ld16(gA + kn + 8, &As[1 - cur][lr * 32 + lc + 8]);
            async_ld16(gB + kn,     &Bs[1 - cur][lr * 32 + lc]);
            async_ld16(gB + kn + 8, &Bs[1 - cur][lr * 32 + lc + 8]);
            // ASYNCcnt completes in order: <=4 outstanding means tile i has
            // landed while tile i+1 keeps streaming behind the WMMAs.
            __builtin_amdgcn_s_wait_asynccnt(4);
        } else {
            __builtin_amdgcn_s_wait_asynccnt(0);
        }
        __syncthreads();
#else
        const int cur = 0;
        const int k0 = i * 32;
        __syncthreads();
        {
            const uint4* ga = (const uint4*)(gA + k0);
            uint4* sa = (uint4*)&As[0][lr * 32 + lc];
            sa[0] = ga[0];
            sa[1] = ga[1];
            const uint4* gb = (const uint4*)(gB + k0);
            uint4* sb = (uint4*)&Bs[0][lr * 32 + lc];
            sb[0] = gb[0];
            sb[1] = gb[1];
        }
        __syncthreads();
#endif
        const unsigned short* Asc = &As[cur][0];
        const unsigned short* Bsc = &Bs[cur][0];

        // A fragments (16x32): lane holds row l16; halves h -> k=(h<8?h:h+8)+8*hi
        Frag a[2];
#pragma unroll
        for (int ia = 0; ia < 2; ++ia) {
            const int row = wm * 32 + ia * 16 + l16;
            a[ia].q[0] = *(const uint4*)&Asc[row * 32 + 8 * hi];
            a[ia].q[1] = *(const uint4*)&Asc[row * 32 + 16 + 8 * hi];
        }
        // B fragments (32x16): lane holds col l16; halves h -> c=(h<8?h:h+8)+8*hi
        Frag b[4];
#pragma unroll
        for (int j = 0; j < 4; ++j) {
            const int row = wn * 64 + j * 16 + l16;
            b[j].q[0] = *(const uint4*)&Bsc[row * 32 + 8 * hi];
            b[j].q[1] = *(const uint4*)&Bsc[row * 32 + 16 + 8 * hi];
        }
#pragma unroll
        for (int ia = 0; ia < 2; ++ia)
#pragma unroll
            for (int j = 0; j < 4; ++j)
                acc[ia][j] = wmma_bf16(a[ia], b[j], acc[ia][j]);
    }

    // Epilogue. C layout: VGPR r, lanes 0-15 -> M=r, lanes 16-31 -> M=8+r.
#pragma unroll
    for (int i = 0; i < 2; ++i)
#pragma unroll
        for (int j = 0; j < 4; ++j)
#pragma unroll
            for (int r = 0; r < 8; ++r) {
                const int m = m0 + wm * 32 + i * 16 + r + 8 * hi;
                const int n = n0 + wn * 64 + j * 16 + l16;
                const float v = acc[i][j][r];
                if (MODE == 0) {
                    const int bb = m >> 11, s = m & 2047;   // S = 2048
                    const int h = n >> 6, d = n & 63;       // D = 64
                    dstbf[(((size_t)(bb * 16 + h)) * 2048 + s) * 64 + d] =
                        f32_to_bf16(v);
                } else {
                    dstf[(size_t)m * 1024 + n] = v + bias[n];
                }
            }
}

// ------------------------------ YaRN RoPE ----------------------------------
// In-place over a (B*H, S=2048, D=64) bf16 buffer. ntk_alpha = S/1024 = 2,
// so inv_freq = 10000^{-(2*d2)/128}; yarn = rsqrt(1 + (pos/1024)^2).
__global__ void rope_kernel(unsigned short* __restrict__ qk, int total) {
    int t = blockIdx.x * blockDim.x + threadIdx.x;
    if (t >= total) return;                    // total = BH * 2048 * 32
    const int d2 = t & 31;
    const int s  = (t >> 5) & 2047;
    const int bh = t >> 16;                    // 2048*32 = 2^16
    const size_t base = ((size_t)bh * 2048 + s) * 64 + 2 * d2;
    const float e = bf16_to_f32(qk[base]);
    const float o = bf16_to_f32(qk[base + 1]);
    const float expo = (float)(2 * d2) * (1.0f / 128.0f);
    const float inv_freq = __expf(-expo * 9.2103403719761836f); // ln(1e4)
    const float p = (float)s;
    const float pn = p * (1.0f / 1024.0f);
    const float yr = __frsqrt_rn(1.0f + pn * pn);
    const float th = p * inv_freq * yr;
    float sn, cs;
    __sincosf(th, &sn, &cs);
    qk[base]     = f32_to_bf16(e * cs - o * sn);
    qk[base + 1] = f32_to_bf16(e * sn + o * cs);
}

// --------------------------- Flash attention -------------------------------
// Grid: (B*H = 64, S/128 = 16). Block: 256 threads = 8 waves, each wave owns
// 16 query rows. Loop over 64-key tiles with online softmax. Q/K/V are bf16
// in (B,H,S,D); output is bf16 in (B,S,E).
__global__ __launch_bounds__(256) void flash_attn_kernel(
    const unsigned short* __restrict__ Q,
    const unsigned short* __restrict__ K,
    const unsigned short* __restrict__ V,
    unsigned short* __restrict__ O) {
    __shared__ unsigned short Ks[64 * 64];          // (key, d) row-major
    __shared__ unsigned short Vt[64 * 64];          // (d, key)  = V^T
    __shared__ unsigned short Ps[8 * 16 * 64];      // per-wave P staging

    const int tid  = threadIdx.x;
    const int wid  = tid >> 5;
    const int lane = tid & 31;
    const int l16  = lane & 15;
    const int hi   = lane >> 4;

    const int bh = blockIdx.x;                 // 0..63
    const int q0 = blockIdx.y * 128;
    const int qw = q0 + wid * 16;              // wave's query-row base
    // Wave-uniform copy in an SGPR: forces s_cmp/s_cbranch for the tile-skip
    // and diag decisions so EXEC stays all-ones around every WMMA (the ISA
    // does NOT skip WMMA when EXEC==0; results would be undefined).
    const int qwu = __builtin_amdgcn_readfirstlane(qw);
    const size_t qkvb = (size_t)bh * 2048 * 64;

    // Q A-fragments straight from global (two K=32 frags covering d=0..63).
    Frag qa[2];
    {
        const unsigned short* qp = Q + qkvb + (size_t)(qw + l16) * 64;
        qa[0].q[0] = *(const uint4*)(qp + 8 * hi);
        qa[0].q[1] = *(const uint4*)(qp + 16 + 8 * hi);
        qa[1].q[0] = *(const uint4*)(qp + 32 + 8 * hi);
        qa[1].q[1] = *(const uint4*)(qp + 48 + 8 * hi);
    }

    v8f oacc[4];
    float mrow[8], lrow[8];
#pragma unroll
    for (int dt = 0; dt < 4; ++dt)
#pragma unroll
        for (int e = 0; e < 8; ++e) oacc[dt][e] = 0.0f;
#pragma unroll
    for (int r = 0; r < 8; ++r) { mrow[r] = -3.0e38f; lrow[r] = 0.0f; }

    const int nkt = (q0 >> 6) + 2;             // key tiles covering q0+127
    const float scale = 0.125f;                // 1/sqrt(64)
    unsigned short* pw = Ps + wid * (16 * 64);

    for (int kt = 0; kt < nkt; ++kt) {
        const int kb = kt * 64;
        __syncthreads();
        // Cooperative load: K tile row-major (async DMA to LDS when
        // available), V tile transposed into LDS through VGPRs.
        {
            const int kk = tid >> 2;               // 0..63 key row
            const int db = (tid & 3) * 16;         // 16-half d chunk
#if USE_ASYNC_LDS
            async_ld16(K + qkvb + (size_t)(kb + kk) * 64 + db,     &Ks[kk * 64 + db]);
            async_ld16(K + qkvb + (size_t)(kb + kk) * 64 + db + 8, &Ks[kk * 64 + db + 8]);
#else
            const uint4* gk = (const uint4*)(K + qkvb + (size_t)(kb + kk) * 64 + db);
            uint4* sk = (uint4*)&Ks[kk * 64 + db];
            sk[0] = gk[0];
            sk[1] = gk[1];
#endif
            Frag vv;
            const uint4* gv = (const uint4*)(V + qkvb + (size_t)(kb + kk) * 64 + db);
            vv.q[0] = gv[0];
            vv.q[1] = gv[1];
#pragma unroll
            for (int i = 0; i < 16; ++i) Vt[(db + i) * 64 + kk] = vv.h[i];
#if USE_ASYNC_LDS
            __builtin_amdgcn_s_wait_asynccnt(0);
#endif
        }
        __syncthreads();

        // Scalar-branch skip of fully-masked tiles (EXEC stays all-ones).
        if (kb <= qwu + 15) {
            // --- scores: S = Q K^T over 64 keys (4 N-tiles x 2 WMMA) ---
            v8f sc[4];
#pragma unroll
            for (int nt = 0; nt < 4; ++nt) {
#pragma unroll
                for (int e = 0; e < 8; ++e) sc[nt][e] = 0.0f;
                Frag b0, b1;
                const int row = nt * 16 + l16;
                b0.q[0] = *(const uint4*)&Ks[row * 64 + 8 * hi];
                b0.q[1] = *(const uint4*)&Ks[row * 64 + 16 + 8 * hi];
                b1.q[0] = *(const uint4*)&Ks[row * 64 + 32 + 8 * hi];
                b1.q[1] = *(const uint4*)&Ks[row * 64 + 48 + 8 * hi];
                sc[nt] = wmma_bf16(qa[0], b0, sc[nt]);
                sc[nt] = wmma_bf16(qa[1], b1, sc[nt]);
            }
            // --- scale + causal mask. 'diag' is a scalar branch: keep the
            // common (fully unmasked) path free of v_cmp/v_cndmask. ---
            if (kb + 63 > qwu) {
#pragma unroll
                for (int nt = 0; nt < 4; ++nt) {
                    const int n = kb + nt * 16 + l16;
#pragma unroll
                    for (int r = 0; r < 8; ++r) {
                        float v = sc[nt][r] * scale;
                        if (n > qw + r + 8 * hi) v = -3.0e38f;
                        sc[nt][r] = v;
                    }
                }
            } else {
#pragma unroll
                for (int nt = 0; nt < 4; ++nt)
#pragma unroll
                    for (int r = 0; r < 8; ++r) sc[nt][r] *= scale;
            }
            // --- online softmax: row max / rescale / row sum (ds_swizzle
            // xor reductions within each 16-lane half) ---
            float alpha[8];
#pragma unroll
            for (int r = 0; r < 8; ++r) {
                float nm = fmaxf(fmaxf(sc[0][r], sc[1][r]),
                                 fmaxf(sc[2][r], sc[3][r]));
                nm = half_reduce_max(nm);
                const float newm = fmaxf(mrow[r], nm);
                alpha[r] = __expf(mrow[r] - newm);
                mrow[r] = newm;
            }
#pragma unroll
            for (int r = 0; r < 8; ++r) {
                float rs = 0.0f;
#pragma unroll
                for (int nt = 0; nt < 4; ++nt) {
                    const float p = __expf(sc[nt][r] - mrow[r]);
                    sc[nt][r] = p;
                    rs += p;
                }
                rs = half_reduce_sum(rs);
                lrow[r] = lrow[r] * alpha[r] + rs;
            }
#pragma unroll
            for (int dt = 0; dt < 4; ++dt)
#pragma unroll
                for (int r = 0; r < 8; ++r) oacc[dt][r] *= alpha[r];
            // --- P (bf16) -> per-wave LDS, reload in A-fragment layout ---
#pragma unroll
            for (int nt = 0; nt < 4; ++nt)
#pragma unroll
                for (int r = 0; r < 8; ++r)
                    pw[(r + 8 * hi) * 64 + nt * 16 + l16] =
                        f32_to_bf16(sc[nt][r]);
            // Same-wave DS ops are in-order; compiler inserts s_wait_dscnt.
            Frag pa[2];
            const unsigned short* pr = pw + (size_t)l16 * 64;
            pa[0].q[0] = *(const uint4*)(pr + 8 * hi);
            pa[0].q[1] = *(const uint4*)(pr + 16 + 8 * hi);
            pa[1].q[0] = *(const uint4*)(pr + 32 + 8 * hi);
            pa[1].q[1] = *(const uint4*)(pr + 48 + 8 * hi);
            // --- O += P V (B-frags contiguous thanks to V^T in LDS) ---
#pragma unroll
            for (int dt = 0; dt < 4; ++dt) {
                Frag v0, v1;
                const int row = dt * 16 + l16;
                v0.q[0] = *(const uint4*)&Vt[row * 64 + 8 * hi];
                v0.q[1] = *(const uint4*)&Vt[row * 64 + 16 + 8 * hi];
                v1.q[0] = *(const uint4*)&Vt[row * 64 + 32 + 8 * hi];
                v1.q[1] = *(const uint4*)&Vt[row * 64 + 48 + 8 * hi];
                oacc[dt] = wmma_bf16(pa[0], v0, oacc[dt]);
                oacc[dt] = wmma_bf16(pa[1], v1, oacc[dt]);
            }
        }
    }

    // Finalize: divide by l, write bf16 into (B, S, E).
    const int b = bh >> 4, h = bh & 15;
#pragma unroll
    for (int dt = 0; dt < 4; ++dt)
#pragma unroll
        for (int r = 0; r < 8; ++r) {
            const int m = qw + r + 8 * hi;
            const int col = h * 64 + dt * 16 + l16;
            O[((size_t)(b * 2048 + m)) * 1024 + col] =
                f32_to_bf16(oacc[dt][r] / lrow[r]);
        }
}

// ------------------------------ launcher -----------------------------------
extern "C" void kernel_launch(void* const* d_in, const int* in_sizes, int n_in,
                              void* d_out, int out_size, void* d_ws,
                              size_t ws_size, hipStream_t stream) {
    const float* x  = (const float*)d_in[0];
    const float* Wq = (const float*)d_in[1];
    const float* Wk = (const float*)d_in[2];
    const float* Wv = (const float*)d_in[3];
    const float* Wp = (const float*)d_in[4];
    const float* bp = (const float*)d_in[5];
    float* out = (float*)d_out;

    // Workspace layout (bytes).
    const size_t XN = 4ull * 2048 * 1024;       // 8,388,608 elems
    const size_t WN = 1024ull * 1024;
    char* ws = (char*)d_ws;
    unsigned short* xbf = (unsigned short*)(ws);
    unsigned short* wqb = (unsigned short*)(ws + 16777216);
    unsigned short* wkb = (unsigned short*)(ws + 16777216 + 1 * 2097152);
    unsigned short* wvb = (unsigned short*)(ws + 16777216 + 2 * 2097152);
    unsigned short* wpb = (unsigned short*)(ws + 16777216 + 3 * 2097152);
    unsigned short* Qb  = (unsigned short*)(ws + 25165824);
    unsigned short* Kb  = (unsigned short*)(ws + 41943040);
    unsigned short* Vb  = (unsigned short*)(ws + 58720256);
    unsigned short* Ab  = (unsigned short*)(ws + 75497472);  // attn out (B,S,E)

    // 1) f32 -> bf16 conversions.
    cvt_f32_bf16_kernel<<<(int)((XN + 255) / 256), 256, 0, stream>>>(x, xbf, (int)XN);
    cvt_f32_bf16_kernel<<<(int)((WN + 255) / 256), 256, 0, stream>>>(Wq, wqb, (int)WN);
    cvt_f32_bf16_kernel<<<(int)((WN + 255) / 256), 256, 0, stream>>>(Wk, wkb, (int)WN);
    cvt_f32_bf16_kernel<<<(int)((WN + 255) / 256), 256, 0, stream>>>(Wv, wvb, (int)WN);
    cvt_f32_bf16_kernel<<<(int)((WN + 255) / 256), 256, 0, stream>>>(Wp, wpb, (int)WN);

    // 2) QKV projections (bf16 WMMA), scattered into (B,H,S,D).
    dim3 ggrid(8, 64);
    gemm_bf16_kernel<0><<<ggrid, 256, 0, stream>>>(xbf, wqb, Qb, nullptr, nullptr);
    gemm_bf16_kernel<0><<<ggrid, 256, 0, stream>>>(xbf, wkb, Kb, nullptr, nullptr);
    gemm_bf16_kernel<0><<<ggrid, 256, 0, stream>>>(xbf, wvb, Vb, nullptr, nullptr);

    // 3) YaRN RoPE on Q and K (in place).
    const int rt = 64 * 2048 * 32;              // BH * S * D/2
    rope_kernel<<<rt / 256, 256, 0, stream>>>(Qb, rt);
    rope_kernel<<<rt / 256, 256, 0, stream>>>(Kb, rt);

    // 4) Causal flash attention -> (B,S,E) bf16.
    flash_attn_kernel<<<dim3(64, 16), 256, 0, stream>>>(Qb, Kb, Vb, Ab);

    // 5) Output projection + bias -> f32 d_out.
    gemm_bf16_kernel<1><<<ggrid, 256, 0, stream>>>(Ab, wpb, nullptr, out, bp);
}
